// Attention_44100724195820
// MI455X (gfx1250) — matrix operand order repaired and checked by reading, not compiled, not run on hardware
//
#include <hip/hip_runtime.h>
#include <hip/hip_bf16.h>

typedef __attribute__((ext_vector_type(16))) _Float16 v16h;
typedef __attribute__((ext_vector_type(8)))  float    v8f;

#define ALPHA 0.2f
#define D_DIM 512   // feature dim of graph_node / virtual_node
#define F_DIM 256   // output feature dim (W columns)
#define C_DIM 64    // number of virtual nodes (softmax width)

// ---------------------------------------------------------------------------
// Kernel 1: setup.
//   blocks 0..63 : vh[c][f] = dot(virtual_node[c], W[:,f])   (also f16 vh^T)
//   block 64     : u[d] = dot(W[d,:], a1)   (the W@a1 folding trick)
// ---------------------------------------------------------------------------
__global__ __launch_bounds__(256) void gat_setup(
    const float* __restrict__ vn, const float* __restrict__ W,
    const float* __restrict__ a, float* __restrict__ vh,
    _Float16* __restrict__ vhT, float* __restrict__ u) {
  const int b = blockIdx.x;
  const int t = threadIdx.x;
  if (b < C_DIM) {
    // vh[b][t]: 512-deep dot; W accessed column-wise -> coalesced across threads
    float acc = 0.f;
    const float* vrow = vn + b * D_DIM;
    #pragma unroll 8
    for (int d = 0; d < D_DIM; ++d)
      acc = fmaf(vrow[d], W[d * F_DIM + t], acc);
    vh[b * F_DIM + t] = acc;
    vhT[t * C_DIM + b] = (_Float16)acc;   // f16 transpose for WMMA B operand
  } else {
    // u[d] = W[d,:] . a1
    for (int d = t; d < D_DIM; d += 256) {
      float acc = 0.f;
      const float* wrow = W + d * F_DIM;
      #pragma unroll 8
      for (int f = 0; f < F_DIM; ++f)
        acc = fmaf(wrow[f], a[f], acc);
      u[d] = acc;
    }
  }
}

// ---------------------------------------------------------------------------
// Kernel 2: s[c] = dot(vh[c,:], a2)  (needs vh from kernel 1; same stream)
// ---------------------------------------------------------------------------
__global__ void gat_svec(const float* __restrict__ vh, const float* __restrict__ a,
                         float* __restrict__ s) {
  const int c = threadIdx.x;   // 64 threads
  float acc = 0.f;
  #pragma unroll 8
  for (int f = 0; f < F_DIM; ++f)
    acc = fmaf(vh[c * F_DIM + f], a[F_DIM + f], acc);
  s[c] = acc;
}

// ---------------------------------------------------------------------------
// Kernel 3: main streaming kernel. One 16-row tile per wave:
//   t = gn_row . u  ->  leakyrelu/softmax over 64 cols  ->  att(16x64 f16)
//   h' tile (16x256 f32) = two V_WMMA_F32_16X16X32_F16 per 16x16 output tile,
//   B operands software-pipelined from LDS.
// ---------------------------------------------------------------------------
__global__ __launch_bounds__(256) void gat_main(
    const float* __restrict__ gn, const _Float16* __restrict__ gVhT,
    const float* __restrict__ gU, const float* __restrict__ gS,
    float* __restrict__ out, int N, int nTiles) {
  __shared__ __align__(32) _Float16 sVhT[F_DIM * C_DIM];  // 32 KB vh^T (f16)
  __shared__ float sU[D_DIM];                             // 2 KB
  __shared__ float sS[C_DIM];                             // 256 B

  // Cooperative LDS fill with b128 copies (vh^T stays hot in 192MB L2)
  {
    const uint4* src = (const uint4*)gVhT;
    uint4* dst = (uint4*)sVhT;
    for (int i = threadIdx.x; i < (F_DIM * C_DIM) / 8; i += 256) dst[i] = src[i];
    for (int i = threadIdx.x; i < D_DIM; i += 256) sU[i] = gU[i];
    if (threadIdx.x < C_DIM) sS[threadIdx.x] = gS[threadIdx.x];
  }
  __syncthreads();

  const int lane = threadIdx.x & 31;
  // Force wave-uniform values into SGPRs: tile/rowBase become scalar, so the
  // tail check is a single s_cbranch and the hot store path has no exec games.
  const int warp = __builtin_amdgcn_readfirstlane(threadIdx.x >> 5);
  const int h = lane >> 4;      // half-wave (selects K/col ranges per ISA layouts)
  const int r = lane & 15;      // row within tile (A/M) and column (B,D/N)

  const int waveId = blockIdx.x * 8 + warp;
  const int waveStride = gridDim.x * 8;

  for (int tile = waveId; tile < nTiles; tile += waveStride) {
    const int rowBase = tile * 16;
    int row = rowBase + r;
    if (row >= N) row = N - 1;   // load clamp (never triggers: 100000 % 16 == 0)

    // ---- Phase A: t[row] = dot(graph_node[row], u); lane pair splits 512 ----
    const float4* gp = (const float4*)(gn + (size_t)row * D_DIM + h * 256);
    const float4* up = (const float4*)(sU + h * 256);
    float t = 0.f;
    #pragma unroll 8
    for (int k = 0; k < 64; ++k) {
      float4 g = gp[k];
      float4 uu = up[k];
      t = fmaf(g.x, uu.x, t);
      t = fmaf(g.y, uu.y, t);
      t = fmaf(g.z, uu.z, t);
      t = fmaf(g.w, uu.w, t);
    }
    t += __shfl_xor(t, 16, 32);  // both halves now hold full row dot

    // ---- Phase B: e = leakyrelu(t + s[c]); softmax over c=0..63.
    // i -> c mapping matches B's contiguous K order:
    //   c = 32*(i>>4) + 16*h + (i&15)
    float e[32];
    float m = -3.0e38f;
    #pragma unroll
    for (int i = 0; i < 32; ++i) {
      const int c = 32 * (i >> 4) + 16 * h + (i & 15);
      float v = t + sS[c];
      v = v > 0.f ? v : ALPHA * v;
      e[i] = v;
      m = fmaxf(m, v);
    }
    m = fmaxf(m, __shfl_xor(m, 16, 32));     // row max across both halves
    float sum = 0.f;
    #pragma unroll
    for (int i = 0; i < 32; ++i) {
      e[i] = __expf(e[i] - m);
      sum += e[i];
    }
    sum += __shfl_xor(sum, 16, 32);          // row sum across both halves
    const float inv = 1.0f / sum;

    v16h A0, A1;                              // att rows, K=0..31 and K=32..63
    #pragma unroll
    for (int i = 0; i < 16; ++i) {
      A0[i] = (_Float16)(e[i] * inv);
      A1[i] = (_Float16)(e[i + 16] * inv);
    }

    // ---- Phase C: 16 output tiles (16x16 f32), K=64 via two WMMAs.
    // B operand: column n = r of vh (f16 vh^T in LDS): K = 16*h + 0..15
    // (+32 for the second K chunk) -> contiguous 32B per lane.
    // Software-pipeline next tile's B loads over WMMA + stores.
    const _Float16* bbase = sVhT + (r * C_DIM + 16 * h);
    v16h B0 = *(const v16h*)(bbase);
    v16h B1 = *(const v16h*)(bbase + 32);

    if (rowBase + 15 < N) {
      // --- full tile: clean store clause, no per-lane guards ---
      #pragma unroll
      for (int nt = 0; nt < 16; ++nt) {
        v16h nB0 = B0, nB1 = B1;
        if (nt < 15) {
          const _Float16* nb = bbase + (nt + 1) * 16 * C_DIM;
          nB0 = *(const v16h*)(nb);
          nB1 = *(const v16h*)(nb + 32);
        }
        v8f acc = {};
        acc = __builtin_amdgcn_wmma_f32_16x16x32_f16(
            false, A0, false, B0, (short)0, acc, false, false);
        acc = __builtin_amdgcn_wmma_f32_16x16x32_f16(
            false, A1, false, B1, (short)0, acc, false, false);

        // D layout: VGPR v -> M = v + 8*h, N = r
        float* op = out + (size_t)(rowBase + 8 * h) * F_DIM + nt * 16 + r;
        #pragma unroll
        for (int v = 0; v < 8; ++v) op[(size_t)v * F_DIM] = acc[v];
        B0 = nB0;
        B1 = nB1;
      }
    } else {
      // --- tail tile (unused for N=100000): per-row guarded stores ---
      #pragma unroll
      for (int nt = 0; nt < 16; ++nt) {
        v16h nB0 = B0, nB1 = B1;
        if (nt < 15) {
          const _Float16* nb = bbase + (nt + 1) * 16 * C_DIM;
          nB0 = *(const v16h*)(nb);
          nB1 = *(const v16h*)(nb + 32);
        }
        v8f acc = {};
        acc = __builtin_amdgcn_wmma_f32_16x16x32_f16(
            false, A0, false, B0, (short)0, acc, false, false);
        acc = __builtin_amdgcn_wmma_f32_16x16x32_f16(
            false, A1, false, B1, (short)0, acc, false, false);

        float* op = out + (size_t)(rowBase + 8 * h) * F_DIM + nt * 16 + r;
        #pragma unroll
        for (int v = 0; v < 8; ++v)
          if (rowBase + 8 * h + v < N) op[(size_t)v * F_DIM] = acc[v];
        B0 = nB0;
        B1 = nB1;
      }
    }
  }
}

// ---------------------------------------------------------------------------
extern "C" void kernel_launch(void* const* d_in, const int* in_sizes, int n_in,
                              void* d_out, int out_size, void* d_ws, size_t ws_size,
                              hipStream_t stream) {
  const float* gn = (const float*)d_in[0];   // graph_node   (N, 512)
  const float* vn = (const float*)d_in[1];   // virtual_node (64, 512)
  const float* W  = (const float*)d_in[2];   // (512, 256)
  const float* a  = (const float*)d_in[3];   // (512, 1)
  float* out = (float*)d_out;                // (N, 256)

  const int N = in_sizes[0] / D_DIM;

  // Workspace layout (~100.5 KB total)
  char* ws = (char*)d_ws;
  float*     vh  = (float*)ws;                                  // 64*256*4 = 65536 B
  _Float16*  vhT = (_Float16*)(ws + 65536);                     // 256*64*2 = 32768 B
  float*     u   = (float*)(ws + 65536 + 32768);                // 512*4    =  2048 B
  float*     s   = (float*)(ws + 65536 + 32768 + 2048);         // 64*4     =   256 B

  gat_setup<<<C_DIM + 1, 256, 0, stream>>>(vn, W, a, vh, vhT, u);
  gat_svec<<<1, C_DIM, 0, stream>>>(vh, a, s);

  const int nTiles = (N + 15) / 16;          // 6250 for N=100000 (exact)
  int blocks = (nTiles + 7) / 8;             // one 16-row tile per wave
  if (blocks > 1024) blocks = 1024;          // stride loop covers the rest
  gat_main<<<blocks, 256, 0, stream>>>(gn, vhT, u, s, out, N, nTiles);
}
